// MambaAggregation_22393959481417
// MI455X (gfx1250) — matrix operand order repaired and checked by reading, not compiled, not run on hardware
//
#include <hip/hip_runtime.h>
#include <hip/hip_bf16.h>
#include <math.h>

typedef __attribute__((ext_vector_type(16))) __bf16 v16bf;
typedef __attribute__((ext_vector_type(8)))  __bf16 v8bf;
typedef __attribute__((ext_vector_type(4)))  __bf16 v4bf;
typedef __attribute__((ext_vector_type(8)))  float  v8f;

#define D   256
#define KNN 8

// ---------------------------------------------------------------------------
// Kernel 0: f32 -> bf16 weight pre-conversion (run once per weight matrix).
// ---------------------------------------------------------------------------
__global__ __launch_bounds__(256) void cvt_bf16_kernel(
    const float* __restrict__ src, __bf16* __restrict__ dst, int n)
{
  int i = (blockIdx.x * 256 + threadIdx.x) * 4;
  if (i + 3 < n) {
    float4 f = *(const float4*)(src + i);
    v4bf h;
    h[0] = (__bf16)f.x; h[1] = (__bf16)f.y;
    h[2] = (__bf16)f.z; h[3] = (__bf16)f.w;
    *(v4bf*)(dst + i) = h;
  } else {
    for (int j = i; j < n; ++j) dst[j] = (__bf16)src[j];
  }
}

// ---------------------------------------------------------------------------
// Kernel 1: brute-force kNN top-8. One thread per query; sp_coords streamed
// through LDS tiles; sorted top-8 kept in registers (unrolled bubble insert).
// ---------------------------------------------------------------------------
#define KTILE 1024
__global__ __launch_bounds__(256) void knn_kernel(
    const float* __restrict__ qpos, const float* __restrict__ spos,
    int* __restrict__ knn_out, int Nq, int N)
{
  __shared__ float sx[KTILE], sy[KTILE], sz[KTILE];
  const int tid = threadIdx.x;
  const int q = blockIdx.x * 256 + tid;
  float qx = 0.f, qy = 0.f, qz = 0.f;
  if (q < Nq) { qx = qpos[3*q]; qy = qpos[3*q+1]; qz = qpos[3*q+2]; }
  float bd[KNN]; int bi[KNN];
  #pragma unroll
  for (int j = 0; j < KNN; ++j) { bd[j] = 3.0e38f; bi[j] = 0; }

  for (int base = 0; base < N; base += KTILE) {
    __syncthreads();
    #pragma unroll
    for (int i = 0; i < KTILE/256; ++i) {
      int p = tid + i*256;
      int gp = base + p;
      if (gp < N) { sx[p] = spos[3*gp]; sy[p] = spos[3*gp+1]; sz[p] = spos[3*gp+2]; }
      else        { sx[p] = 1.0e30f;    sy[p] = 1.0e30f;      sz[p] = 1.0e30f; }
    }
    __syncthreads();
    for (int t = 0; t < KTILE; ++t) {
      float dx = qx - sx[t], dy = qy - sy[t], dz = qz - sz[t];
      float d2 = fmaf(dx, dx, fmaf(dy, dy, dz*dz));
      if (d2 < bd[KNN-1]) {
        bd[KNN-1] = d2; bi[KNN-1] = base + t;
        #pragma unroll
        for (int j = KNN-1; j > 0; --j) {
          if (bd[j] < bd[j-1]) {
            float td = bd[j]; bd[j] = bd[j-1]; bd[j-1] = td;
            int   ti = bi[j]; bi[j] = bi[j-1]; bi[j-1] = ti;
          }
        }
      }
    }
  }
  if (q < Nq) {
    #pragma unroll
    for (int j = 0; j < KNN; ++j) knn_out[q*KNN + j] = bi[j];
  }
}

// ---------------------------------------------------------------------------
// Kernel 2/4/5: C[M,256] = (A[M,256] @ Wb[256,256]^T) [* mul] [+ res] [LN].
// A is f32 (converted to bf16 once at LDS staging); Wb is pre-converted bf16.
// 128 threads = 4 wave32s; block computes 16 full output rows so LayerNorm
// is done in-block. Inner loop: 2x ds_load_b128 + 2x global_load_b128 + wmma.
// ---------------------------------------------------------------------------
__global__ __launch_bounds__(128) void gemm256_wmma(
    const float* __restrict__ A, const __bf16* __restrict__ Wb,
    const float* __restrict__ mul, const float* __restrict__ res,
    const float* __restrict__ gamma, const float* __restrict__ beta,
    float* __restrict__ C, int M)
{
  __shared__ __bf16 As[16][264];  // bf16 A tile; stride mult. of 8 -> 16B runs
  __shared__ float  Os[16][258];  // f32 output staging for epilogue/LN

  const int tid  = threadIdx.x;
  const int lane = tid & 31;
  const int wave = tid >> 5;
  const int m0   = blockIdx.x * 16;

  // stage A tile (16 x 256) into LDS as bf16; convert once per element
  #pragma unroll
  for (int i = 0; i < 8; ++i) {
    int lin4 = tid + i*128;        // float4 index
    int r = lin4 >> 6;             // (lin4*4)/256
    int c = (lin4 & 63) * 4;
    int gr = m0 + r; if (gr >= M) gr = M - 1;
    float4 f = *(const float4*)(A + (size_t)gr * D + c);
    v4bf h;
    h[0] = (__bf16)f.x; h[1] = (__bf16)f.y;
    h[2] = (__bf16)f.z; h[3] = (__bf16)f.w;
    *(v4bf*)&As[r][c] = h;
  }
  __syncthreads();

  const int half = lane >> 4;       // which half of the wave
  const int mrow = lane & 15;       // A row for this lane
  // each wave handles 4 of the 16 N-tiles
  for (int t = 0; t < 4; ++t) {
    int n0 = (wave*4 + t) * 16;
    int ncol = n0 + (lane & 15);
    const __bf16* wrow = Wb + (size_t)ncol * D;
    __builtin_prefetch(wrow, 0, 0);   // global_prefetch_b8
    v8f acc = {0.f,0.f,0.f,0.f,0.f,0.f,0.f,0.f};
    #pragma unroll
    for (int k0 = 0; k0 < D; k0 += 32) {
      // A 16x32 bf16 layout: per lane, K k0+16*half .. +15: one contiguous
      // 16-element run (matches B's k order).
      v8bf alo = *(const v8bf*)&As[mrow][k0 + 16*half];
      v8bf ahi = *(const v8bf*)&As[mrow][k0 + 16*half + 8];
      v16bf a = __builtin_shufflevector(alo, ahi,
          0,1,2,3,4,5,6,7,8,9,10,11,12,13,14,15);
      // B 32x16 bf16 layout: per lane (N=ncol), K k0+16*half .. +15:
      // one contiguous 16-element run of W[ncol][.] (B[k][n] = W[n][k]).
      const __bf16* bp = wrow + k0 + 16*half;
      v8bf blo = *(const v8bf*)(bp);
      v8bf bhi = *(const v8bf*)(bp + 8);
      v16bf b = __builtin_shufflevector(blo, bhi,
          0,1,2,3,4,5,6,7,8,9,10,11,12,13,14,15);
      acc = __builtin_amdgcn_wmma_f32_16x16x32_bf16(
          /*neg_a=*/false, a, /*neg_b=*/false, b,
          /*c_mod=*/(short)0, acc, /*reuse_a=*/false, /*reuse_b=*/false);
    }
    // C/D layout: VGPR j -> row j (lanes 0-15) / row j+8 (lanes 16-31)
    #pragma unroll
    for (int j = 0; j < 8; ++j) {
      int mm = j + 8*half;
      int nn = n0 + (lane & 15);
      float v = acc[j];
      size_t g = (size_t)(m0 + mm) * D + nn;
      if (mul) v *= mul[g];
      if (res) v += res[g];
      Os[mm][nn] = v;
    }
  }
  __syncthreads();

  if (gamma) {
    // LayerNorm: 8 threads per row, 32 columns each; reduce in 8-lane groups
    int row = tid >> 3, sub = tid & 7;
    float s = 0.f, s2 = 0.f;
    #pragma unroll
    for (int j = 0; j < 32; ++j) {
      float v = Os[row][sub*32 + j];
      s += v; s2 = fmaf(v, v, s2);
    }
    #pragma unroll
    for (int msk = 4; msk >= 1; msk >>= 1) {
      s  += __shfl_xor(s,  msk, 32);
      s2 += __shfl_xor(s2, msk, 32);
    }
    float mean = s * (1.0f / D);
    float var  = s2 * (1.0f / D) - mean * mean;
    float rs   = rsqrtf(var + 1e-5f);
    int gr = m0 + row;
    if (gr < M) {
      #pragma unroll
      for (int j = 0; j < 32; ++j) {
        int c = sub*32 + j;
        C[(size_t)gr * D + c] = (Os[row][c] - mean) * rs * gamma[c] + beta[c];
      }
    }
  } else {
    #pragma unroll
    for (int i = 0; i < 32; ++i) {
      int lin = tid + i*128;
      int r = lin >> 8, c = lin & 255;
      int gr = m0 + r;
      if (gr < M) C[(size_t)gr * D + c] = Os[r][c];
    }
  }
}

// ---------------------------------------------------------------------------
// Kernel 3: gate = softmax(qf @ w_k^T + w_b); fbar = sum_k gate_k * feat[idx_k].
// One wave per query.
// ---------------------------------------------------------------------------
__global__ __launch_bounds__(256) void gate_fbar_kernel(
    const float* __restrict__ qf, const float* __restrict__ feats,
    const int* __restrict__ knn, const float* __restrict__ wk,
    const float* __restrict__ wb, float* __restrict__ fbar, int Nq)
{
  const int tid  = threadIdx.x;
  const int lane = tid & 31;
  const int wave = tid >> 5;
  const int q = blockIdx.x * 8 + wave;
  if (q >= Nq) return;

  float qv[8];
  #pragma unroll
  for (int j = 0; j < 8; ++j) qv[j] = qf[(size_t)q * D + lane + 32*j];

  float logit[KNN];
  #pragma unroll
  for (int k = 0; k < KNN; ++k) {
    float p = 0.f;
    #pragma unroll
    for (int j = 0; j < 8; ++j) p = fmaf(qv[j], wk[k*D + lane + 32*j], p);
    #pragma unroll
    for (int m = 16; m >= 1; m >>= 1) p += __shfl_xor(p, m, 32);
    logit[k] = p + wb[k];
  }
  float mx = logit[0];
  #pragma unroll
  for (int k = 1; k < KNN; ++k) mx = fmaxf(mx, logit[k]);
  float g[KNN]; float s = 0.f;
  #pragma unroll
  for (int k = 0; k < KNN; ++k) { g[k] = __expf(logit[k] - mx); s += g[k]; }
  float inv = 1.0f / s;
  int idx[KNN];
  #pragma unroll
  for (int k = 0; k < KNN; ++k) { idx[k] = knn[q*KNN + k]; g[k] *= inv; }

  #pragma unroll
  for (int j = 0; j < 8; ++j) {
    int dcol = lane + 32*j;
    float acc = 0.f;
    #pragma unroll
    for (int k = 0; k < KNN; ++k)
      acc = fmaf(g[k], feats[(size_t)idx[k] * D + dcol], acc);
    fbar[(size_t)q * D + dcol] = acc;
  }
}

// ---------------------------------------------------------------------------
extern "C" void kernel_launch(void* const* d_in, const int* in_sizes, int n_in,
                              void* d_out, int out_size, void* d_ws, size_t ws_size,
                              hipStream_t stream) {
  const float* query = (const float*)d_in[0];
  const float* qpos  = (const float*)d_in[1];
  const float* feats = (const float*)d_in[2];
  const float* spos  = (const float*)d_in[3];
  const float* w_q   = (const float*)d_in[4];
  const float* w_v   = (const float*)d_in[5];
  const float* w_o   = (const float*)d_in[6];
  const float* w_k   = (const float*)d_in[7];
  const float* w_b   = (const float*)d_in[8];
  const float* gamma = (const float*)d_in[9];
  const float* beta  = (const float*)d_in[10];
  float* out = (float*)d_out;

  const int Nq = in_sizes[1] / 3;
  const int N  = in_sizes[3] / 3;

  auto align256 = [](size_t x) { return (x + 255) & ~(size_t)255; };
  char* ws = (char*)d_ws;
  size_t off = 0;
  int*    knn_idx = (int*)(ws + off);    off += align256((size_t)Nq * KNN * sizeof(int));
  float*  qf      = (float*)(ws + off);  off += align256((size_t)Nq * D * sizeof(float));
  float*  fbar    = (float*)(ws + off);  off += align256((size_t)Nq * D * sizeof(float));
  float*  vbarw   = (float*)(ws + off);  off += align256((size_t)Nq * D * sizeof(float));
  __bf16* wq_bf   = (__bf16*)(ws + off); off += align256((size_t)D * D * sizeof(__bf16));
  __bf16* wv_bf   = (__bf16*)(ws + off); off += align256((size_t)D * D * sizeof(__bf16));
  __bf16* wo_bf   = (__bf16*)(ws + off); off += align256((size_t)D * D * sizeof(__bf16));
  (void)ws_size; (void)n_in; (void)out_size;

  const int nW = D * D;
  const int cvtBlocks = (nW / 4 + 255) / 256;
  // 0) weights -> bf16 (once per call; cheap)
  cvt_bf16_kernel<<<cvtBlocks, 256, 0, stream>>>(w_q, wq_bf, nW);
  cvt_bf16_kernel<<<cvtBlocks, 256, 0, stream>>>(w_v, wv_bf, nW);
  cvt_bf16_kernel<<<cvtBlocks, 256, 0, stream>>>(w_o, wo_bf, nW);
  // 1) kNN indices
  knn_kernel<<<(Nq + 255) / 256, 256, 0, stream>>>(qpos, spos, knn_idx, Nq, N);
  // 2) qf = query @ w_q^T
  gemm256_wmma<<<(Nq + 15) / 16, 128, 0, stream>>>(
      query, wq_bf, nullptr, nullptr, nullptr, nullptr, qf, Nq);
  // 3) gate + fbar = sum_k gate_k * feat_k
  gate_fbar_kernel<<<(Nq + 7) / 8, 256, 0, stream>>>(
      qf, feats, knn_idx, w_k, w_b, fbar, Nq);
  // 4) vbarw = (fbar @ w_v^T) ⊙ qf
  gemm256_wmma<<<(Nq + 15) / 16, 128, 0, stream>>>(
      fbar, wv_bf, qf, nullptr, nullptr, nullptr, vbarw, Nq);
  // 5) out = LN(vbarw @ w_o^T + query) * gamma + beta
  gemm256_wmma<<<(Nq + 15) / 16, 128, 0, stream>>>(
      vbarw, wo_bf, nullptr, query, gamma, beta, out, Nq);
}
